// Qwen2VisionModelBase_73924977099242
// MI455X (gfx1250) — compile-verified, matrix operand order checked by reading, not run on hardware
//
#include <hip/hip_runtime.h>
#include <math.h>

// ---- model constants (t=1, h=48, w=48 per setup_inputs) ----
#define S_TOK   2304
#define HIDDEN_ 1280
#define HEADS_  16
#define HDIM    80
#define DPAD    96      // head_dim padded for 3x k=32 WMMA steps
#define INTER_  3420
#define DEPTH_  4
#define MERGED  576     // S / (MERGE*MERGE)
#define NWIN    36      // 6x6 full windows of 64 tokens

typedef __attribute__((ext_vector_type(16))) __bf16 v16bf;
typedef __attribute__((ext_vector_type(8)))  float  v8f;
typedef unsigned int u32x4 __attribute__((ext_vector_type(4)));

union FragBF { u32x4 q[2]; v16bf v; };

__device__ __forceinline__ unsigned short f2bf(float f) {
  union { float f; unsigned int u; } c; c.f = f;
  unsigned int u = c.u + 0x7fffu + ((c.u >> 16) & 1u);   // round-to-nearest-even
  return (unsigned short)(u >> 16);
}

// window-order token j -> original token index (grid 48x48, merge 2, vmw 4)
__device__ __forceinline__ int winPermToken(int j) {
  int u = j & 3, g = j >> 2;
  int c = g & 3, r = (g >> 2) & 3, b = (g >> 4) % 6, a = g / 96;
  int hh = 4 * a + r, ww = 4 * b + c;
  return ((hh * 24 + ww) << 2) + u;
}
// window-order merged row g -> original merged row (rev permutation)
__device__ __forceinline__ int mergeRevRow(int g) {
  int c = g & 3, r = (g >> 2) & 3, b = (g >> 4) % 6, a = g / 96;
  return (4 * a + r) * 24 + (4 * b + c);
}

// branch-free stage of 16 consecutive f32 -> bf16 into LDS (caller guarantees bounds)
__device__ __forceinline__ void stageFull(const float* __restrict__ src,
                                          unsigned short* __restrict__ dst) {
#pragma unroll
  for (int j = 0; j < 16; j += 4) {
    float4 v4 = *(const float4*)(src + j);
    dst[j + 0] = f2bf(v4.x); dst[j + 1] = f2bf(v4.y);
    dst[j + 2] = f2bf(v4.z); dst[j + 3] = f2bf(v4.w);
  }
}

// ---------------- RoPE cos/sin tables (window order) ----------------
__global__ void rope_tables(float* __restrict__ cosT, float* __restrict__ sinT) {
  int idx = blockIdx.x * blockDim.x + threadIdx.x;
  if (idx >= S_TOK * 40) return;
  int s = idx / 40, d = idx % 40;
  int i = winPermToken(s);
  int wm = i & 1, hm = (i >> 1) & 1, m = i >> 2;
  int ww = m % 24, hh = m / 24;
  int pos = (d < 20) ? (2 * hh + hm) : (2 * ww + wm);
  int f = d % 20;
  float inv = powf(10000.f, -(float)f / 20.f);
  float ang = (float)pos * inv;
  cosT[idx] = cosf(ang);
  sinT[idx] = sinf(ang);
}

// ---------------- RMSNorm over 1280 cols, f32 out ----------------
__global__ __launch_bounds__(256) void rms_norm(const float* __restrict__ x,
                                                const float* __restrict__ w,
                                                float* __restrict__ out) {
  __shared__ float red[8];
  int row = blockIdx.x;
  const float* xr = x + (size_t)row * HIDDEN_;
  float s = 0.f;
  for (int c = threadIdx.x; c < HIDDEN_; c += 256) { float v = xr[c]; s += v * v; }
  for (int o = 16; o > 0; o >>= 1) s += __shfl_down(s, o, 32);
  if ((threadIdx.x & 31) == 0) red[threadIdx.x >> 5] = s;
  __syncthreads();
  if (threadIdx.x == 0) {
    float t = 0.f;
    for (int i = 0; i < 8; ++i) t += red[i];
    red[0] = rsqrtf(t / (float)HIDDEN_ + 1e-6f);
  }
  __syncthreads();
  float rs = red[0];
  float* orow = out + (size_t)row * HIDDEN_;
  for (int c = threadIdx.x; c < HIDDEN_; c += 256) orow[c] = xr[c] * rs * w[c];
}

// ---------------- QKV rope/split: f32 (S,3840) -> bf16 (H,S,96) ----------------
__global__ void qkv_rope(const float* __restrict__ qkv,
                         const float* __restrict__ cosT,
                         const float* __restrict__ sinT,
                         unsigned short* __restrict__ qb,
                         unsigned short* __restrict__ kb,
                         unsigned short* __restrict__ vb) {
  int s = blockIdx.x, h = blockIdx.y, d = threadIdx.x;     // d in 0..95
  size_t ob = ((size_t)h * S_TOK + s) * DPAD + d;
  if (d >= HDIM) { qb[ob] = 0; kb[ob] = 0; vb[ob] = 0; return; }
  const float* base = qkv + (size_t)s * 3 * HIDDEN_ + h * HDIM;
  float qv = base[d], kv = base[HIDDEN_ + d], vv = base[2 * HIDDEN_ + d];
  float cv = cosT[s * 40 + (d % 40)];
  float sv = sinT[s * 40 + (d % 40)];
  float qo = (d < 40) ? -base[d + 40] : base[d - 40];
  float ko = (d < 40) ? -base[HIDDEN_ + d + 40] : base[HIDDEN_ + d - 40];
  const float scale = 0.111803398874989485f;               // 80^-0.5 folded into q
  qb[ob] = f2bf((qv * cv + qo * sv) * scale);
  kb[ob] = f2bf(kv * cv + ko * sv);
  vb[ob] = f2bf(vv);
}

// ---------------- generic bf16 WMMA GEMM: out = act(A @ W^T + bias [+ res]) ----------------
#define BM 128
#define BN 128
#define BK 32
#define AST 40   // LDS row stride (bf16 elems): 80B -> b128-aligned fragment loads

__global__ __launch_bounds__(256) void gemm_bf16(
    const float* __restrict__ A, int lda, int a_perm,
    const float* __restrict__ W, int b_layout,   // 0: W is (N,K); 1: W is (K,N), N%BN==0
    const float* __restrict__ bias,
    const float* __restrict__ res,
    float* __restrict__ out, int out_perm,
    int M, int N, int K, int act)                // act: 0 none, 1 exact GELU
{
  __shared__ unsigned short As[BM * AST];
  __shared__ unsigned short Bs[BN * AST];
  int tid  = threadIdx.x;
  int wave = tid >> 5, lane = tid & 31;
  int lg = lane >> 4, ln = lane & 15;
  int wm = wave >> 1, wn = wave & 1;
  int m0 = blockIdx.y * BM, n0 = blockIdx.x * BN;

  v8f acc[2][4];
#pragma unroll
  for (int i = 0; i < 2; ++i)
#pragma unroll
    for (int j = 0; j < 4; ++j)
#pragma unroll
      for (int e = 0; e < 8; ++e) acc[i][j][e] = 0.f;

  int lrowA = tid >> 1;            // 0..127
  int lcolA = (tid & 1) * 16;      // 0 or 16
  // Row clamping: OOB rows stage garbage from a valid row; those C rows/cols are
  // masked in the epilogue, so no zero-fill is needed (only K needs true zeros).
  int agr = m0 + lrowA;
  int agrc = (agr < M) ? agr : (M - 1);
  int asrc = a_perm ? winPermToken(agrc) : agrc;
  int bgr = n0 + lrowA;            // layout 0
  int bsrcRow = (bgr < N) ? bgr : (N - 1);
  int bkk = tid >> 3;              // 0..31  (layout 1)
  int bns = (tid & 7) * 16;        // 0..112

  const float* aRow = A + (size_t)asrc * lda + lcolA;
  const float* bRow = W + (size_t)bsrcRow * K + lcolA;   // layout 0 only
  unsigned short* aDst = &As[lrowA * AST + lcolA];
  unsigned short* bDst = &Bs[lrowA * AST + lcolA];

  auto mma_step = [&]() {
    FragBF af[2], bfr[4];
#pragma unroll
    for (int mt = 0; mt < 2; ++mt) {
      const unsigned short* p = &As[(wm * 32 + mt * 16 + ln) * AST + lg * 8];
      af[mt].q[0] = *(const u32x4*)p;
      af[mt].q[1] = *(const u32x4*)(p + 16);
    }
#pragma unroll
    for (int nt = 0; nt < 4; ++nt) {
      const unsigned short* p = &Bs[(wn * 64 + nt * 16 + ln) * AST + lg * 8];
      bfr[nt].q[0] = *(const u32x4*)p;
      bfr[nt].q[1] = *(const u32x4*)(p + 16);
    }
#pragma unroll
    for (int mt = 0; mt < 2; ++mt)
#pragma unroll
      for (int nt = 0; nt < 4; ++nt)
        acc[mt][nt] = __builtin_amdgcn_wmma_f32_16x16x32_bf16(
            false, af[mt].v, false, bfr[nt].v, (short)0, acc[mt][nt], false, false);
  };

  int kfull = K & ~(BK - 1);
  // ---- main loop: staging fully in-bounds in K, completely branch-free ----
  for (int k0 = 0; k0 < kfull; k0 += BK) {
    stageFull(aRow + k0, aDst);
    if (b_layout == 0) {
      stageFull(bRow + k0, bDst);
    } else {
      const float* srcW = W + (size_t)(k0 + bkk) * N + n0 + bns;
#pragma unroll
      for (int j = 0; j < 16; j += 4) {
        float4 v4 = *(const float4*)(srcW + j);
        Bs[(bns + j + 0) * AST + bkk] = f2bf(v4.x);
        Bs[(bns + j + 1) * AST + bkk] = f2bf(v4.y);
        Bs[(bns + j + 2) * AST + bkk] = f2bf(v4.z);
        Bs[(bns + j + 3) * AST + bkk] = f2bf(v4.w);
      }
    }
    __syncthreads();
    mma_step();
    __syncthreads();
  }
  // ---- K tail (runs at most once; K%32 != 0 only for K=1176, 3420) ----
  if (kfull < K) {
    const float* arow0 = A + (size_t)asrc * lda;
#pragma unroll
    for (int j = 0; j < 16; ++j) {
      int k = kfull + lcolA + j;
      bool in = (k < K);
      float v = arow0[in ? k : 0];
      aDst[j] = in ? f2bf(v) : (unsigned short)0;
    }
    if (b_layout == 0) {
      const float* brow0 = W + (size_t)bsrcRow * K;
#pragma unroll
      for (int j = 0; j < 16; ++j) {
        int k = kfull + lcolA + j;
        bool in = (k < K);
        float v = brow0[in ? k : 0];
        bDst[j] = in ? f2bf(v) : (unsigned short)0;
      }
    } else {
      int k = kfull + bkk;
      bool kin = (k < K);
      const float* srcW = W + (size_t)(kin ? k : 0) * N + n0 + bns;
#pragma unroll
      for (int j = 0; j < 16; j += 4) {
        float4 v4 = *(const float4*)(srcW + j);
        Bs[(bns + j + 0) * AST + bkk] = kin ? f2bf(v4.x) : (unsigned short)0;
        Bs[(bns + j + 1) * AST + bkk] = kin ? f2bf(v4.y) : (unsigned short)0;
        Bs[(bns + j + 2) * AST + bkk] = kin ? f2bf(v4.z) : (unsigned short)0;
        Bs[(bns + j + 3) * AST + bkk] = kin ? f2bf(v4.w) : (unsigned short)0;
      }
    }
    __syncthreads();
    mma_step();
    __syncthreads();
  }

  // ---- epilogue ----
#pragma unroll
  for (int mt = 0; mt < 2; ++mt) {
#pragma unroll
    for (int nt = 0; nt < 4; ++nt) {
      int gn = n0 + wn * 64 + nt * 16 + ln;
      if (gn >= N) continue;
      float bv = bias ? bias[gn] : 0.f;
#pragma unroll
      for (int r = 0; r < 8; ++r) {
        int gm = m0 + wm * 32 + mt * 16 + r + 8 * lg;
        if (gm >= M) continue;
        float v = acc[mt][nt][r] + bv;
        if (res) v += res[(size_t)gm * N + gn];
        if (act == 1) v = 0.5f * v * (1.f + erff(v * 0.70710678118654752f));
        int om = out_perm ? mergeRevRow(gm) : gm;
        out[(size_t)om * N + gn] = v;
      }
    }
  }
}

// ---------------- flash attention (windowed or full), bf16 WMMA ----------------
#define KST 104
#define VST 72
#define PST 72

__global__ __launch_bounds__(128) void flash_attn(
    const unsigned short* __restrict__ qb,
    const unsigned short* __restrict__ kb,
    const unsigned short* __restrict__ vb,
    float* __restrict__ out, int nkt, int windowed)
{
  __shared__ unsigned short Ks[64 * KST];
  __shared__ unsigned short Vt[96 * VST];      // V transposed: [dim][key]
  __shared__ unsigned short Ps[4][16 * PST];   // per-wave P tile
  int h = blockIdx.y;
  int qblk = blockIdx.x;
  int wave = threadIdx.x >> 5;
  int lane = threadIdx.x & 31;
  int lg = lane >> 4, ln = lane & 15;
  int q0 = qblk * 64 + wave * 16;

  FragBF qf[3];
  {
    const unsigned short* qrow = qb + ((size_t)h * S_TOK + q0 + ln) * DPAD;
#pragma unroll
    for (int kk = 0; kk < 3; ++kk) {
      qf[kk].q[0] = *(const u32x4*)(qrow + kk * 32 + lg * 8);
      qf[kk].q[1] = *(const u32x4*)(qrow + kk * 32 + 16 + lg * 8);
    }
  }
  v8f O[6];
  float mrow[8], lrow[8];
#pragma unroll
  for (int t = 0; t < 6; ++t)
#pragma unroll
    for (int e = 0; e < 8; ++e) O[t][e] = 0.f;
#pragma unroll
  for (int r = 0; r < 8; ++r) { mrow[r] = -1e30f; lrow[r] = 0.f; }

  for (int kt = 0; kt < nkt; ++kt) {
    int kbase = windowed ? qblk * 64 : kt * 64;
    {   // cooperative K/V staging: 2 threads per key row
      int t = threadIdx.x;
      int row = t >> 1, c0 = (t & 1) * 48;
      const unsigned short* ksrc = kb + ((size_t)h * S_TOK + kbase + row) * DPAD + c0;
#pragma unroll
      for (int j = 0; j < 6; ++j)
        *(u32x4*)(&Ks[row * KST + c0 + j * 8]) = *(const u32x4*)(ksrc + j * 8);
      const unsigned short* vsrc = vb + ((size_t)h * S_TOK + kbase + row) * DPAD + c0;
#pragma unroll
      for (int j = 0; j < 48; ++j)
        Vt[(c0 + j) * VST + row] = vsrc[j];
    }
    __syncthreads();

    // S = Q K^T  (16 queries x 64 keys), 12 WMMA
    v8f sc[4];
#pragma unroll
    for (int nt = 0; nt < 4; ++nt)
#pragma unroll
      for (int e = 0; e < 8; ++e) sc[nt][e] = 0.f;
#pragma unroll
    for (int kk = 0; kk < 3; ++kk) {
#pragma unroll
      for (int nt = 0; nt < 4; ++nt) {
        FragBF bk;
        const unsigned short* pK = &Ks[(nt * 16 + ln) * KST + kk * 32 + lg * 8];
        bk.q[0] = *(const u32x4*)pK;
        bk.q[1] = *(const u32x4*)(pK + 16);
        sc[nt] = __builtin_amdgcn_wmma_f32_16x16x32_bf16(
            false, qf[kk].v, false, bk.v, (short)0, sc[nt], false, false);
      }
    }
    // online softmax (row m = r + 8*lg, reduce across the 16-lane half)
#pragma unroll
    for (int r = 0; r < 8; ++r) {
      float tmax = fmaxf(fmaxf(sc[0][r], sc[1][r]), fmaxf(sc[2][r], sc[3][r]));
#pragma unroll
      for (int off = 8; off; off >>= 1) tmax = fmaxf(tmax, __shfl_xor(tmax, off, 32));
      float mn = fmaxf(mrow[r], tmax);
      float corr = __expf(mrow[r] - mn);
      float rs = 0.f;
#pragma unroll
      for (int nt = 0; nt < 4; ++nt) {
        float pv = __expf(sc[nt][r] - mn);
        sc[nt][r] = pv; rs += pv;
      }
#pragma unroll
      for (int off = 8; off; off >>= 1) rs += __shfl_xor(rs, off, 32);
      lrow[r] = lrow[r] * corr + rs;
      mrow[r] = mn;
#pragma unroll
      for (int t = 0; t < 6; ++t) O[t][r] *= corr;
    }
    // P -> LDS (C-layout -> row major), then reload as A fragments
#pragma unroll
    for (int nt = 0; nt < 4; ++nt)
#pragma unroll
      for (int r = 0; r < 8; ++r)
        Ps[wave][(r + 8 * lg) * PST + nt * 16 + ln] = f2bf(sc[nt][r]);
    FragBF pf[2];
#pragma unroll
    for (int kk = 0; kk < 2; ++kk) {
      const unsigned short* pp = &Ps[wave][ln * PST + kk * 32 + lg * 8];
      pf[kk].q[0] = *(const u32x4*)pp;
      pf[kk].q[1] = *(const u32x4*)(pp + 16);
    }
    // O += P V  (16 x 96), 12 WMMA
#pragma unroll
    for (int nt = 0; nt < 6; ++nt) {
#pragma unroll
      for (int kk = 0; kk < 2; ++kk) {
        FragBF vf;
        const unsigned short* pv = &Vt[(nt * 16 + ln) * VST + kk * 32 + lg * 8];
        vf.q[0] = *(const u32x4*)pv;
        vf.q[1] = *(const u32x4*)(pv + 16);
        O[nt] = __builtin_amdgcn_wmma_f32_16x16x32_bf16(
            false, pf[kk].v, false, vf.v, (short)0, O[nt], false, false);
      }
    }
    __syncthreads();
  }
#pragma unroll
  for (int nt = 0; nt < 6; ++nt) {
    int d = nt * 16 + ln;
    if (d < HDIM) {
#pragma unroll
      for (int r = 0; r < 8; ++r) {
        int srow = q0 + r + 8 * lg;
        out[(size_t)srow * HIDDEN_ + h * HDIM + d] = O[nt][r] / lrow[r];
      }
    }
  }
}

// ---------------- SiLU(g) * u, in place on g ----------------
__global__ void silu_mul(float* __restrict__ g, const float* __restrict__ u, int n) {
  int i = blockIdx.x * blockDim.x + threadIdx.x;
  if (i < n) {
    float gv = g[i];
    g[i] = (gv / (1.f + __expf(-gv))) * u[i];
  }
}

// =====================================================================
extern "C" void kernel_launch(void* const* d_in, const int* in_sizes, int n_in,
                              void* d_out, int out_size, void* d_ws, size_t ws_size,
                              hipStream_t stream)
{
  (void)in_sizes; (void)n_in; (void)out_size; (void)ws_size;
  const float* hidden  = (const float*)d_in[0];
  const float* patch_w = (const float*)d_in[1];
  const float* norm1   = (const float*)d_in[2];
  const float* qkv_w   = (const float*)d_in[3];
  const float* qkv_b   = (const float*)d_in[4];
  const float* o_w     = (const float*)d_in[5];
  const float* o_b     = (const float*)d_in[6];
  const float* norm2   = (const float*)d_in[7];
  const float* gate_w  = (const float*)d_in[8];
  const float* gate_b  = (const float*)d_in[9];
  const float* up_w    = (const float*)d_in[10];
  const float* up_b    = (const float*)d_in[11];
  const float* down_w  = (const float*)d_in[12];
  const float* down_b  = (const float*)d_in[13];
  const float* lnq_w   = (const float*)d_in[14];
  const float* fc1_w   = (const float*)d_in[15];
  const float* fc1_b   = (const float*)d_in[16];
  const float* fc2_w   = (const float*)d_in[17];
  const float* fc2_b   = (const float*)d_in[18];

  // ---- carve workspace (~168 MB total) ----
  char* p = (char*)d_ws;
  auto carve = [&](size_t bytes) -> void* {
    void* r = (void*)p;
    p += (bytes + 255) & ~(size_t)255;
    return r;
  };
  float* x    = (float*)carve((size_t)S_TOK * HIDDEN_ * 4);
  float* xn   = (float*)carve((size_t)S_TOK * HIDDEN_ * 4);   // also merger view (576,5120)
  float* qkv  = (float*)carve((size_t)S_TOK * 3 * HIDDEN_ * 4);
  unsigned short* qb = (unsigned short*)carve((size_t)HEADS_ * S_TOK * DPAD * 2);
  unsigned short* kb = (unsigned short*)carve((size_t)HEADS_ * S_TOK * DPAD * 2);
  unsigned short* vb = (unsigned short*)carve((size_t)HEADS_ * S_TOK * DPAD * 2);
  float* attn = (float*)carve((size_t)S_TOK * HIDDEN_ * 4);
  float* G    = (float*)carve((size_t)S_TOK * INTER_ * 4);
  float* U    = (float*)carve((size_t)S_TOK * INTER_ * 4);
  float* Hbuf = (float*)carve((size_t)MERGED * 5120 * 4);
  float* cosT = (float*)carve((size_t)S_TOK * 40 * 4);
  float* sinT = (float*)carve((size_t)S_TOK * 40 * 4);

  auto gemm = [&](const float* A, int lda, int a_perm,
                  const float* W, int b_layout, const float* bias,
                  const float* res, float* out, int out_perm,
                  int M, int N, int K, int act) {
    dim3 grid((N + BN - 1) / BN, (M + BM - 1) / BM);
    gemm_bf16<<<grid, 256, 0, stream>>>(A, lda, a_perm, W, b_layout, bias, res,
                                        out, out_perm, M, N, K, act);
  };

  rope_tables<<<(S_TOK * 40 + 255) / 256, 256, 0, stream>>>(cosT, sinT);

  // patch embed: x(window order) = hidden[perm] @ patch_w   (patch_w is (K,N), N=1280)
  gemm(hidden, 1176, 1, patch_w, 1, nullptr, nullptr, x, 0, S_TOK, HIDDEN_, 1176, 0);

  for (int i = 0; i < DEPTH_; ++i) {
    rms_norm<<<S_TOK, 256, 0, stream>>>(x, norm1 + (size_t)i * HIDDEN_, xn);
    gemm(xn, HIDDEN_, 0, qkv_w + (size_t)i * 3 * HIDDEN_ * HIDDEN_, 0,
         qkv_b + (size_t)i * 3 * HIDDEN_, nullptr, qkv, 0,
         S_TOK, 3 * HIDDEN_, HIDDEN_, 0);
    qkv_rope<<<dim3(S_TOK, HEADS_), 96, 0, stream>>>(qkv, cosT, sinT, qb, kb, vb);
    int full = (i == 3);
    flash_attn<<<dim3(NWIN, HEADS_), 128, 0, stream>>>(qb, kb, vb, attn,
                                                       full ? NWIN : 1, full ? 0 : 1);
    gemm(attn, HIDDEN_, 0, o_w + (size_t)i * HIDDEN_ * HIDDEN_, 0,
         o_b + (size_t)i * HIDDEN_, x, x, 0, S_TOK, HIDDEN_, HIDDEN_, 0);

    rms_norm<<<S_TOK, 256, 0, stream>>>(x, norm2 + (size_t)i * HIDDEN_, xn);
    gemm(xn, HIDDEN_, 0, gate_w + (size_t)i * INTER_ * HIDDEN_, 0,
         gate_b + (size_t)i * INTER_, nullptr, G, 0, S_TOK, INTER_, HIDDEN_, 0);
    gemm(xn, HIDDEN_, 0, up_w + (size_t)i * INTER_ * HIDDEN_, 0,
         up_b + (size_t)i * INTER_, nullptr, U, 0, S_TOK, INTER_, HIDDEN_, 0);
    int nel = S_TOK * INTER_;
    silu_mul<<<(nel + 255) / 256, 256, 0, stream>>>(G, U, nel);
    gemm(G, INTER_, 0, down_w + (size_t)i * HIDDEN_ * INTER_, 0,
         down_b + (size_t)i * HIDDEN_, x, x, 0, S_TOK, HIDDEN_, INTER_, 0);
  }

  // merger: rms -> (576,5120) -> fc1+gelu -> fc2 (+row un-permute into d_out)
  rms_norm<<<S_TOK, 256, 0, stream>>>(x, lnq_w, xn);
  gemm(xn, 5120, 0, fc1_w, 0, fc1_b, nullptr, Hbuf, 0, MERGED, 5120, 5120, 1);
  gemm(Hbuf, 5120, 0, fc2_w, 0, fc2_b, nullptr, (float*)d_out, 1, MERGED, 2048, 5120, 0);
}
